// GINEncoder_29291676959176
// MI455X (gfx1250) — matrix operand order repaired and checked by reading, not compiled, not run on hardware
//
#include <hip/hip_runtime.h>

#define DIM 128

typedef __attribute__((ext_vector_type(16))) _Float16 v16h;
typedef __attribute__((ext_vector_type(2)))  __fp16   fp16x2;
typedef __attribute__((ext_vector_type(8)))  float    v8f;

// ---------------- utility kernels ----------------

__global__ void zero_kernel(float* p, int n) {
    int i = blockIdx.x * blockDim.x + threadIdx.x;
    if (i < n) p[i] = 0.0f;
}

// convert fp32 weights to f16 AND transpose to [n][k] so that B-fragment
// reads from LDS are contiguous 32-byte runs.
__global__ void convert_weights_kernel(const float* __restrict__ W1,
                                       const float* __restrict__ W2,
                                       _Float16* __restrict__ Wh,
                                       int perMat, int nLayers) {
    int i = blockIdx.x * blockDim.x + threadIdx.x;
    int total = perMat * nLayers;
    if (i >= total) return;
    int l = i / perMat, r = i % perMat;
    int k = r / DIM, n = r % DIM;            // source is [k][n]
    size_t t = (size_t)n * DIM + k;          // dest is [n][k]
    Wh[(size_t)(l * 2 + 0) * perMat + t] = (_Float16)W1[i];
    Wh[(size_t)(l * 2 + 1) * perMat + t] = (_Float16)W2[i];
}

// one wave (32 lanes) per edge; float4 gather + 4 float atomics per lane
__global__ void scatter_add_kernel(const float* __restrict__ h,
                                   const long long* __restrict__ ei,
                                   float* __restrict__ agg, int nEdges) {
    int warp = (blockIdx.x * blockDim.x + threadIdx.x) >> 5;
    int lane = threadIdx.x & 31;
    if (warp >= nEdges) return;
    int s = (int)ei[warp];
    int d = (int)ei[nEdges + warp];
    const float4 v = ((const float4*)(h + (size_t)s * DIM))[lane];
    float* dp = agg + (size_t)d * DIM + lane * 4;
    atomicAdd(dp + 0, v.x);
    atomicAdd(dp + 1, v.y);
    atomicAdd(dp + 2, v.z);
    atomicAdd(dp + 3, v.w);
}

// ---------------- WMMA GEMM ----------------
// MODE 0: A = h + agg              (GIN aggregation add, fused)
// MODE 1: A = relu(scale*z+shift)  (BatchNorm + ReLU, fused)
// Epilogue: bias add + optional per-column sum/sumsq atomics for BN stats.

template <int MODE>
__global__ __launch_bounds__(256)
void gemm_wmma_kernel(const float* __restrict__ A0,
                      const float* __restrict__ A1,    // MODE0: agg
                      const float* __restrict__ scsh,  // MODE1: [scale|shift]
                      const _Float16* __restrict__ W,  // [n][k] f16 (transposed)
                      const float* __restrict__ bias,
                      float* __restrict__ out,
                      float* __restrict__ stats,       // [sum|sumsq]
                      int doStats, int mTiles) {
    __shared__ _Float16 Wlds[DIM * DIM];   // 32 KB weight panel, [n][k]
    {
        const uint4* src = (const uint4*)W;
        uint4* dst = (uint4*)Wlds;
        for (int i = threadIdx.x; i < (DIM * DIM) / 8; i += blockDim.x)
            dst[i] = src[i];
    }
    __syncthreads();

    int wid   = threadIdx.x >> 5;
    int lane  = threadIdx.x & 31;
    int mtile = blockIdx.x * 8 + wid;
    if (mtile >= mTiles) return;

    int half_id = lane >> 4;
    int l15     = lane & 15;
    int row     = mtile * 16 + l15;
    const float* arow0 = A0 + (size_t)row * DIM;
    const float* arow1 = (MODE == 0) ? (A1 + (size_t)row * DIM) : A0;

    v8f acc[8] = {};

    #pragma unroll
    for (int chunk = 0; chunk < 4; chunk++) {
        int kbase = chunk * 32;

        // --- A fragment (contiguous k = 16h + s order, same as B) ---
        // lane half h: elements 0-7  -> K = kbase + 16h + [0..7]
        //              elements 8-15 -> K = kbase + 16h + 8 + [0..7]
        // Each run is 8 contiguous floats: vector loads + pk_rtz packing.
        union { v16h v; fp16x2 h2[8]; } A;
        #pragma unroll
        for (int run = 0; run < 2; run++) {
            int k = kbase + 16 * half_id + run * 8;
            float4 v0 = *(const float4*)(arow0 + k);
            float4 v1 = *(const float4*)(arow0 + k + 4);
            float vals[8] = {v0.x, v0.y, v0.z, v0.w, v1.x, v1.y, v1.z, v1.w};
            if (MODE == 0) {
                float4 g0 = *(const float4*)(arow1 + k);
                float4 g1 = *(const float4*)(arow1 + k + 4);
                float gs[8] = {g0.x, g0.y, g0.z, g0.w, g1.x, g1.y, g1.z, g1.w};
                #pragma unroll
                for (int j = 0; j < 8; j++) vals[j] += gs[j];
            } else {
                float4 s0 = *(const float4*)(scsh + k);
                float4 s1 = *(const float4*)(scsh + k + 4);
                float4 t0 = *(const float4*)(scsh + DIM + k);
                float4 t1 = *(const float4*)(scsh + DIM + k + 4);
                float sc[8] = {s0.x, s0.y, s0.z, s0.w, s1.x, s1.y, s1.z, s1.w};
                float sh[8] = {t0.x, t0.y, t0.z, t0.w, t1.x, t1.y, t1.z, t1.w};
                #pragma unroll
                for (int j = 0; j < 8; j++) {
                    float t = sc[j] * vals[j] + sh[j];
                    vals[j] = t > 0.f ? t : 0.f;
                }
            }
            #pragma unroll
            for (int j = 0; j < 4; j++)
                A.h2[run * 4 + j] =
                    __builtin_amdgcn_cvt_pkrtz(vals[2 * j], vals[2 * j + 1]);
        }

        // --- 8 N-tiles; B fragment = 16 contiguous halves from [n][k] LDS ---
        #pragma unroll
        for (int nt = 0; nt < 8; nt++) {
            // element i -> K = kbase + 16*half + i, N = nt*16 + l15
            const v16h b = *(const v16h*)(
                &Wlds[(size_t)(nt * 16 + l15) * DIM + kbase + 16 * half_id]);
            acc[nt] = __builtin_amdgcn_wmma_f32_16x16x32_f16(
                false, A.v, false, b, (short)0, acc[nt], false, false);
        }
    }

    // --- epilogue: bias, store, BN column statistics ---
    #pragma unroll
    for (int nt = 0; nt < 8; nt++) {
        int n = nt * 16 + l15;
        float bv = bias[n];
        float s = 0.f, ss = 0.f;
        #pragma unroll
        for (int r = 0; r < 8; r++) {
            float v = acc[nt][r] + bv;
            out[(size_t)(mtile * 16 + r + 8 * half_id) * DIM + n] = v;
            s += v;
            ss += v * v;
        }
        if (doStats) {
            atomicAdd(&stats[n], s);
            atomicAdd(&stats[DIM + n], ss);
        }
    }
}

// sums -> fused BN scale/shift
__global__ void finalize_bn_kernel(const float* __restrict__ stats,
                                   const float* __restrict__ gamma,
                                   const float* __restrict__ beta,
                                   float* __restrict__ scsh, float invM) {
    int c = threadIdx.x;
    if (c >= DIM) return;
    float mean = stats[c] * invM;
    float var  = stats[DIM + c] * invM - mean * mean;
    float inv  = rsqrtf(var + 1e-5f);
    float sc   = gamma[c] * inv;
    scsh[c]       = sc;
    scsh[DIM + c] = beta[c] - mean * sc;
}

__global__ void bn_relu_kernel(float* __restrict__ h,
                               const float* __restrict__ scsh, int total) {
    int i = blockIdx.x * blockDim.x + threadIdx.x;
    if (i >= total) return;
    int c = i & (DIM - 1);
    float v = h[i] * scsh[c] + scsh[DIM + c];
    h[i] = v > 0.f ? v : 0.f;
}

// one wave per node; shuffle reduction (wave32)
__global__ void l2norm_kernel(const float* __restrict__ h,
                              float* __restrict__ out, int n) {
    int gw   = (blockIdx.x * blockDim.x + threadIdx.x) >> 5;
    int lane = threadIdx.x & 31;
    if (gw >= n) return;
    const float4 v = ((const float4*)(h + (size_t)gw * DIM))[lane];
    float ss = v.x * v.x + v.y * v.y + v.z * v.z + v.w * v.w;
    #pragma unroll
    for (int off = 16; off > 0; off >>= 1) ss += __shfl_xor(ss, off, 32);
    float inv = 1.0f / fmaxf(sqrtf(ss), 1e-12f);
    float4 o = {v.x * inv, v.y * inv, v.z * inv, v.w * inv};
    ((float4*)(out + (size_t)gw * DIM))[lane] = o;
}

// ---------------- launch ----------------

extern "C" void kernel_launch(void* const* d_in, const int* in_sizes, int n_in,
                              void* d_out, int out_size, void* d_ws, size_t ws_size,
                              hipStream_t stream) {
    const float*     x   = (const float*)d_in[0];
    const long long* ei  = (const long long*)d_in[1];  // int64 edge_index [2][E]
    const float*     W1  = (const float*)d_in[2];      // [3][128][128]
    const float*     b1  = (const float*)d_in[3];
    const float*     g1  = (const float*)d_in[4];
    const float*     be1 = (const float*)d_in[5];
    const float*     W2  = (const float*)d_in[6];
    const float*     b2  = (const float*)d_in[7];
    const float*     gbn = (const float*)d_in[8];
    const float*     bbn = (const float*)d_in[9];

    int nNodes = in_sizes[0] / DIM;
    int nEdges = in_sizes[1] / 2;
    int feat   = nNodes * DIM;
    int mTiles = (nNodes + 15) / 16;   // 50000/16 = 3125 exact

    // workspace layout
    float* ws    = (float*)d_ws;
    float* h     = ws;                        // [N,128]
    float* agg   = ws + (size_t)feat;         // [N,128]
    float* z1    = ws + (size_t)2 * feat;     // [N,128]
    float* stats = ws + (size_t)3 * feat;     // 6 regions x 256
    float* scsh  = stats + 6 * 256;           // 6 regions x 256
    _Float16* Wh = (_Float16*)(scsh + 6 * 256);  // 3 layers x 2 mats x 128x128

    const int perMat = DIM * DIM;

    convert_weights_kernel<<<(3 * perMat + 255) / 256, 256, 0, stream>>>(
        W1, W2, Wh, perMat, 3);
    zero_kernel<<<(6 * 256 + 255) / 256, 256, 0, stream>>>(stats, 6 * 256);

    float invM = 1.0f / (float)nNodes;
    int gblocks = (mTiles + 7) / 8;

    for (int i = 0; i < 3; i++) {
        const float* hcur = (i == 0) ? x : h;

        // aggregation: agg = segment_sum(h[src] -> dst)
        zero_kernel<<<(feat + 255) / 256, 256, 0, stream>>>(agg, feat);
        {
            long long thr = (long long)nEdges * 32;
            scatter_add_kernel<<<(int)((thr + 255) / 256), 256, 0, stream>>>(
                hcur, ei, agg, nEdges);
        }

        // GEMM1: z1 = (h + agg) @ W1[i] + b1[i]   (+ column stats for BN1)
        gemm_wmma_kernel<0><<<gblocks, 256, 0, stream>>>(
            hcur, agg, nullptr,
            Wh + (size_t)(i * 2 + 0) * perMat, b1 + i * DIM,
            z1, stats + i * 256, 1, mTiles);
        finalize_bn_kernel<<<1, DIM, 0, stream>>>(
            stats + i * 256, g1 + i * DIM, be1 + i * DIM, scsh + i * 256, invM);

        // GEMM2: h = relu(BN(z1)) @ W2[i] + b2[i] (+ stats for inter-layer BN)
        int doStats2 = (i < 2) ? 1 : 0;
        gemm_wmma_kernel<1><<<gblocks, 256, 0, stream>>>(
            z1, nullptr, scsh + i * 256,
            Wh + (size_t)(i * 2 + 1) * perMat, b2 + i * DIM,
            h, stats + (3 + i) * 256, doStats2, mTiles);

        if (i < 2) {
            finalize_bn_kernel<<<1, DIM, 0, stream>>>(
                stats + (3 + i) * 256, gbn + i * DIM, bbn + i * DIM,
                scsh + (3 + i) * 256, invM);
            bn_relu_kernel<<<(feat + 255) / 256, 256, 0, stream>>>(
                h, scsh + (3 + i) * 256, feat);
        }
    }

    // final row-wise L2 normalize -> d_out
    {
        long long thr = (long long)nNodes * 32;
        l2norm_kernel<<<(int)((thr + 255) / 256), 256, 0, stream>>>(
            h, (float*)d_out, nNodes);
    }
}
